// INT4QuantizedLinear_38482906972513
// MI455X (gfx1250) — matrix operand order repaired and checked by reading, not compiled, not run on hardware
//
#include <hip/hip_runtime.h>

typedef _Float16 half_t;
typedef __attribute__((ext_vector_type(16))) _Float16 v16h;
typedef __attribute__((ext_vector_type(8)))  _Float16 v8h;
typedef __attribute__((ext_vector_type(8)))  float    v8f;
typedef __attribute__((ext_vector_type(4)))  float    v4f;
typedef __attribute__((ext_vector_type(4)))  int      v4i;

#define K_DIM 4096
#define N_DIM 4096
#define KPACK (K_DIM / 2)   // packed "bytes" (one per int32) per output row

// ---------------------------------------------------------------------------
// Preprocess 1: unpack int4 weights -> f16 (exact: values in [-8,7]).
// Each thread: 4 packed int32 "bytes" -> 8 f16 (one b128 store).
// ---------------------------------------------------------------------------
__global__ __launch_bounds__(256) void dequant_w_kernel(
    const int* __restrict__ PW, half_t* __restrict__ Wh)
{
    const int i = (blockIdx.x * 256 + threadIdx.x) * 4;   // packed index
    v4i v = *(const v4i*)(PW + i);
    v8h o;
#pragma unroll
    for (int j = 0; j < 4; ++j) {
        o[2 * j + 0] = (half_t)(((v[j] >> 4) & 15) - 8);  // even flat idx: high nibble
        o[2 * j + 1] = (half_t)(( v[j]       & 15) - 8);  // odd  flat idx: low nibble
    }
    *(v8h*)(Wh + 2 * (size_t)i) = o;
}

// ---------------------------------------------------------------------------
// Preprocess 2: x f32 -> f16. Each thread: 8 floats -> 8 halves.
// ---------------------------------------------------------------------------
__global__ __launch_bounds__(256) void convert_x_kernel(
    const float* __restrict__ X, half_t* __restrict__ Xh)
{
    const size_t i = ((size_t)blockIdx.x * 256 + threadIdx.x) * 8;
    v4f x0 = *(const v4f*)(X + i);
    v4f x1 = *(const v4f*)(X + i + 4);
    v8h o;
#pragma unroll
    for (int j = 0; j < 4; ++j) {
        o[j]     = (half_t)x0[j];
        o[4 + j] = (half_t)x1[j];
    }
    *(v8h*)(Xh + i) = o;
}

// ---------------------------------------------------------------------------
// Main GEMM (fast path): both operands already f16 in workspace.
// Block tile 128x256, 8 waves, each wave a 64x64 tile = 4x4 WMMA accumulators.
// Inner loop: 16 b128 loads + 16 v_wmma_f32_16x16x32_f16, no conversions.
// ---------------------------------------------------------------------------
__global__ __launch_bounds__(256, 1) void gemm_f16_wmma_kernel(
    const half_t* __restrict__ Xh,      // [M, K] f16
    const half_t* __restrict__ Wh,      // [N, K] f16 (B columns = W rows)
    const int*    __restrict__ PB,      // [N/2]
    const float*  __restrict__ scale_w_p,
    const float*  __restrict__ scale_b_p,
    float* __restrict__ Out)            // [M, N]
{
    const int lane  = threadIdx.x & 31;
    const int wave  = threadIdx.x >> 5;
    const int wm    = wave >> 2;        // 0..1
    const int wn    = wave & 3;         // 0..3
    const int l16   = lane & 15;
    const int lhalf = lane >> 4;

    const int m0 = blockIdx.y * 128 + wm * 64;
    const int n0 = blockIdx.x * 256 + wn * 64;

    // A fragment (16x32 f16): lane row m=l16; per lane two 8-half runs at
    // K = k + lhalf*8 and K = k + lhalf*8 + 16.
    const half_t* arow[4];
#pragma unroll
    for (int mi = 0; mi < 4; ++mi)
        arow[mi] = Xh + (size_t)(m0 + mi * 16 + l16) * K_DIM + lhalf * 8;

    // B fragment (32x16 f16): lane col n=l16; per lane two 8-half runs at
    // K = k + lhalf*8 and K = k + lhalf*8 + 16.
    const half_t* brow[4];
#pragma unroll
    for (int ni = 0; ni < 4; ++ni)
        brow[ni] = Wh + (size_t)(n0 + ni * 16 + l16) * K_DIM + lhalf * 8;

    v8f acc[4][4] = {};

    for (int k = 0; k < K_DIM; k += 32) {
        v16h a[4], b[4];
#pragma unroll
        for (int mi = 0; mi < 4; ++mi) {
            v8h lo = *(const v8h*)(arow[mi] + k);
            v8h hi = *(const v8h*)(arow[mi] + k + 16);
            a[mi] = __builtin_shufflevector(lo, hi,
                      0, 1, 2, 3, 4, 5, 6, 7, 8, 9, 10, 11, 12, 13, 14, 15);
        }
#pragma unroll
        for (int ni = 0; ni < 4; ++ni) {
            v8h lo = *(const v8h*)(brow[ni] + k);
            v8h hi = *(const v8h*)(brow[ni] + k + 16);
            b[ni] = __builtin_shufflevector(lo, hi,
                      0, 1, 2, 3, 4, 5, 6, 7, 8, 9, 10, 11, 12, 13, 14, 15);
        }

        if (k + 32 < K_DIM) {
            __builtin_prefetch((const void*)(arow[0] + k + 128), 0, 1);
            __builtin_prefetch((const void*)(brow[0] + k + 128), 0, 1);
        }

#pragma unroll
        for (int mi = 0; mi < 4; ++mi)
#pragma unroll
            for (int ni = 0; ni < 4; ++ni)
                acc[mi][ni] = __builtin_amdgcn_wmma_f32_16x16x32_f16(
                    false, a[mi], false, b[ni],
                    (short)0, acc[mi][ni], false, false);
    }

    const float sw = *scale_w_p;
    const float sb = *scale_b_p;

#pragma unroll
    for (int ni = 0; ni < 4; ++ni) {
        const int n   = n0 + ni * 16 + l16;
        const int pbv = PB[n >> 1];
        const int nib = (n & 1) ? (pbv & 15) : ((pbv >> 4) & 15);
        const float bias = sb * (float)(nib - 8);
#pragma unroll
        for (int mi = 0; mi < 4; ++mi) {
            const int mbase = m0 + mi * 16 + lhalf * 8;
#pragma unroll
            for (int r = 0; r < 8; ++r)
                Out[(size_t)(mbase + r) * N_DIM + n] = sw * acc[mi][ni][r] + bias;
        }
    }
}

// ---------------------------------------------------------------------------
// Fallback (no workspace): fused on-the-fly dequant + convert (round-1 kernel).
// ---------------------------------------------------------------------------
__global__ __launch_bounds__(256, 1) void int4_linear_fused_kernel(
    const float* __restrict__ X,
    const int*   __restrict__ PW,
    const int*   __restrict__ PB,
    const float* __restrict__ scale_w_p,
    const float* __restrict__ scale_b_p,
    float* __restrict__ Out)
{
    const int lane  = threadIdx.x & 31;
    const int wave  = threadIdx.x >> 5;
    const int wm    = wave >> 2;
    const int wn    = wave & 3;
    const int l16   = lane & 15;
    const int lhalf = lane >> 4;

    const int m0 = blockIdx.y * 128 + wm * 64;
    const int n0 = blockIdx.x * 256 + wn * 64;

    const float* arow[4];
#pragma unroll
    for (int mi = 0; mi < 4; ++mi)
        arow[mi] = X + (size_t)(m0 + mi * 16 + l16) * K_DIM + lhalf * 8;

    const int* brow[4];
#pragma unroll
    for (int ni = 0; ni < 4; ++ni)
        brow[ni] = PW + (size_t)(n0 + ni * 16 + l16) * KPACK + lhalf * 4;

    v8f acc[4][4] = {};

    for (int k = 0; k < K_DIM; k += 32) {
        v16h a[4], b[4];
#pragma unroll
        for (int mi = 0; mi < 4; ++mi) {
            const float* p = arow[mi] + k;
            v4f x0 = *(const v4f*)(p + 0);
            v4f x1 = *(const v4f*)(p + 4);
            v4f x2 = *(const v4f*)(p + 16);
            v4f x3 = *(const v4f*)(p + 20);
#pragma unroll
            for (int j = 0; j < 4; ++j) {
                a[mi][j +  0] = (half_t)x0[j];
                a[mi][j +  4] = (half_t)x1[j];
                a[mi][j +  8] = (half_t)x2[j];
                a[mi][j + 12] = (half_t)x3[j];
            }
        }
#pragma unroll
        for (int ni = 0; ni < 4; ++ni) {
            const int* q = brow[ni] + (k >> 1);
            v4i w0 = *(const v4i*)(q + 0);
            v4i w1 = *(const v4i*)(q + 8);
#pragma unroll
            for (int j = 0; j < 4; ++j) {
                b[ni][2 * j + 0] = (half_t)(((w0[j] >> 4) & 15) - 8);
                b[ni][2 * j + 1] = (half_t)(( w0[j]       & 15) - 8);
                b[ni][2 * j + 8] = (half_t)(((w1[j] >> 4) & 15) - 8);
                b[ni][2 * j + 9] = (half_t)(( w1[j]       & 15) - 8);
            }
        }
#pragma unroll
        for (int mi = 0; mi < 4; ++mi)
#pragma unroll
            for (int ni = 0; ni < 4; ++ni)
                acc[mi][ni] = __builtin_amdgcn_wmma_f32_16x16x32_f16(
                    false, a[mi], false, b[ni],
                    (short)0, acc[mi][ni], false, false);
    }

    const float sw = *scale_w_p;
    const float sb = *scale_b_p;

#pragma unroll
    for (int ni = 0; ni < 4; ++ni) {
        const int n   = n0 + ni * 16 + l16;
        const int pbv = PB[n >> 1];
        const int nib = (n & 1) ? (pbv & 15) : ((pbv >> 4) & 15);
        const float bias = sb * (float)(nib - 8);
#pragma unroll
        for (int mi = 0; mi < 4; ++mi) {
            const int mbase = m0 + mi * 16 + lhalf * 8;
#pragma unroll
            for (int r = 0; r < 8; ++r)
                Out[(size_t)(mbase + r) * N_DIM + n] = sw * acc[mi][ni][r] + bias;
        }
    }
}

// ---------------------------------------------------------------------------
extern "C" void kernel_launch(void* const* d_in, const int* in_sizes, int n_in,
                              void* d_out, int out_size, void* d_ws, size_t ws_size,
                              hipStream_t stream) {
    const float* X  = (const float*)d_in[0];   // [4,2048,4096] f32
    const int*   PW = (const int*)d_in[1];     // [8388608] int32 (one packed byte each)
    const int*   PB = (const int*)d_in[2];     // [2048] int32
    const float* sw = (const float*)d_in[3];
    const float* sb = (const float*)d_in[4];
    float* Out = (float*)d_out;

    const int M = in_sizes[0] / K_DIM;         // 8192
    dim3 grid(N_DIM / 256, M / 128);           // (16, 64)

    const size_t wh_bytes = (size_t)N_DIM * K_DIM * sizeof(half_t);   // 32 MiB
    const size_t xh_bytes = (size_t)in_sizes[0] * sizeof(half_t);     // 64 MiB

    if (ws_size >= wh_bytes + xh_bytes) {
        half_t* Wh = (half_t*)d_ws;
        half_t* Xh = (half_t*)((char*)d_ws + wh_bytes);

        // 8,388,608 packed ints / 4 per thread / 256 per block = 8192 blocks
        dequant_w_kernel<<<(N_DIM * KPACK / 4) / 256, 256, 0, stream>>>(PW, Wh);
        // 33,554,432 floats / 8 per thread / 256 per block = 16384 blocks
        convert_x_kernel<<<(in_sizes[0] / 8) / 256, 256, 0, stream>>>(X, Xh);

        gemm_f16_wmma_kernel<<<grid, 256, 0, stream>>>(Xh, Wh, PB, sw, sb, Out);
    } else {
        int4_linear_fused_kernel<<<grid, 256, 0, stream>>>(X, PW, PB, sw, sb, Out);
    }
}
